// SelfAttention_90829968376273
// MI455X (gfx1250) — matrix operand order repaired and checked by reading, not compiled, not run on hardware
//
#include <hip/hip_runtime.h>
#include <hip/hip_bf16.h>
#include <math.h>

// ---------------------------------------------------------------------------
// Problem constants (from the reference)
// ---------------------------------------------------------------------------
#define DIMC     2048
#define NHEADS   16
#define HEADD    128      // concat head dim
#define HALFHD   64
#define QDIM     1024
#define KVDIM    1024
#define TOTALC   3072
#define BB       2
#define SS       2048
#define EPSV     1e-5f

typedef __attribute__((ext_vector_type(16))) _Float16 v16h;
typedef __attribute__((ext_vector_type(8)))  _Float16 v8h;
typedef __attribute__((ext_vector_type(8)))  float    v8f;

// ---------------------------------------------------------------------------
// WMMA fragment helpers (wave32, A/B in this code's contiguous k order; C/D per CDNA5 ISA 7.12.2)
//
// A (16x32 f16):  lanes 0-15 -> row M=lane,   K = k0..k0+15   (contiguous)
//                 lanes16-31 -> row M=lane-16,K = k0+16..k0+31
// B (32x16 f16):  lanes 0-15 -> col N=lane,    K = k0..k0+15   (contiguous)
//                 lanes16-31 -> col N=lane-16, K = k0+16..k0+31
// C/D (16x16 f32): lanes 0-15 -> N=lane, M=r ; lanes16-31 -> N=lane-16, M=8+r
// ---------------------------------------------------------------------------
__device__ __forceinline__ v16h wmma_load_a(const _Float16* A, int lda, int k0) {
    const int lane = threadIdx.x & 31;
    const _Float16* p = A + (size_t)(lane & 15) * lda + k0 + ((lane >> 4) << 4);
    v8h lo = *(const v8h*)p;          // K kbase .. kbase+7
    v8h hi = *(const v8h*)(p + 8);    // K kbase+8 .. kbase+15
    v16h r;
#pragma unroll
    for (int i = 0; i < 8; ++i) { r[i] = lo[i]; r[i + 8] = hi[i]; }
    return r;
}

// B sourced from a row-major [N][K] matrix (operand of X @ W^T):
// each lane reads 16 contiguous f16 from row (n0 + col).
__device__ __forceinline__ v16h wmma_load_bT(const _Float16* W, int ldw, int k0) {
    const int lane = threadIdx.x & 31;
    const _Float16* p = W + (size_t)(lane & 15) * ldw + k0 + ((lane >> 4) << 4);
    return *(const v16h*)p;
}

__device__ __forceinline__ v8f wmma_f16(v16h a, v16h b, v8f c) {
    return __builtin_amdgcn_wmma_f32_16x16x32_f16(false, a, false, b,
                                                  (short)0, c, false, false);
}

// ---------------------------------------------------------------------------
// Async copy: stage a ROWS-row x 32-col f16 tile from global to LDS via
// GLOBAL_LOAD_ASYNC_TO_LDS_B128 (ASYNCcnt-tracked, CDNA5 §08).
// (ROWS/8) issues x 32 lanes x 16B. ldsOff is a raw LDS byte offset
// (the VGPR holds the LDS byte address), valid because the GEMM kernel has no static
// __shared__ (extern dynamic LDS base is 0).
// ---------------------------------------------------------------------------
template<int ROWS>
__device__ __forceinline__ void async_stage_tile(const _Float16* src, int ldk,
                                                 unsigned ldsOff) {
    const int lane = threadIdx.x & 31;
#pragma unroll
    for (int it = 0; it < ROWS / 8; ++it) {
        const int chunk = it * 32 + lane;      // chunks of 16B (8 f16)
        const int row = chunk >> 2;            // 0..ROWS-1
        const int c   = chunk & 3;             // 0..3
        unsigned long long ga =
            (unsigned long long)(uintptr_t)(src + (size_t)row * ldk + c * 8);
        unsigned la = ldsOff + (unsigned)(row * 64 + c * 16);
        asm volatile("global_load_async_to_lds_b128 %0, %1, off"
                     :: "v"(la), "v"(ga) : "memory");
    }
}

#define ASYNC_WAIT_12() asm volatile("s_wait_asynccnt 12" ::: "memory")
#define ASYNC_WAIT_0()  asm volatile("s_wait_asynccnt 0"  ::: "memory")

// ---------------------------------------------------------------------------
// Kernel 0: fp32 -> f16 conversion
// ---------------------------------------------------------------------------
__global__ void k_cvt_f16(const float* __restrict__ in, _Float16* __restrict__ out, int n) {
    int i = blockIdx.x * blockDim.x + threadIdx.x;
    if (i < n) out[i] = (_Float16)in[i];
}

// ---------------------------------------------------------------------------
// Kernel 1: C[M x N] (f32) = A16[M x K] @ W16[N x K]^T
// One wave per 64x32 tile (4x2 WMMA accumulators = 64 acc VGPRs -> no spill).
// Operands double-buffered in LDS via async-to-LDS copies:
// per wave 2 x (4KB A + 2KB B) = 12 KB slice; 4 waves -> 48 KB dynamic LDS.
// No barriers (each wave owns its slice). 12 async issues per k-step.
// ---------------------------------------------------------------------------
__global__ void __launch_bounds__(128, 1)
k_gemm_wmma(const _Float16* __restrict__ A,
            const _Float16* __restrict__ W,
            float* __restrict__ C,
            int M, int N, int K) {
    extern __shared__ char dynsmem[];
    const int waveid = threadIdx.x >> 5;
    const int wave   = blockIdx.x * (blockDim.x >> 5) + waveid;
    const int tn32   = N >> 5;
    const int tiles  = (M >> 6) * tn32;
    if (wave >= tiles) return;
    const int m0 = (wave / tn32) << 6;
    const int n0 = (wave % tn32) << 5;

    const unsigned slice = (unsigned)waveid * 12288u;   // per-wave LDS slice
    const _Float16* Abase = A + (size_t)m0 * K;
    const _Float16* Wbase = W + (size_t)n0 * K;

    v8f acc[4][2];
#pragma unroll
    for (int i = 0; i < 4; ++i)
#pragma unroll
        for (int j = 0; j < 2; ++j) acc[i][j] = (v8f){};

    const int nk = K >> 5;
    // prologue: stage k-step 0 into buffer 0
    async_stage_tile<64>(Abase, K, slice);
    async_stage_tile<32>(Wbase, K, slice + 4096u);

#pragma unroll 1
    for (int kt = 0; kt < nk; ++kt) {
        const int p = kt & 1;
        if (kt + 1 < nk) {
            const unsigned nb = slice + (unsigned)((p ^ 1) * 6144);
            async_stage_tile<64>(Abase + (kt + 1) * 32, K, nb);
            async_stage_tile<32>(Wbase + (kt + 1) * 32, K, nb + 4096u);
            ASYNC_WAIT_12();     // oldest 12 (current buffer) complete, in order
        } else {
            ASYNC_WAIT_0();
        }

        const _Float16* lA = (const _Float16*)(dynsmem + slice + p * 6144);
        const _Float16* lB = lA + 2048;   // +4096 bytes

        v16h b[2];
#pragma unroll
        for (int j = 0; j < 2; ++j) b[j] = wmma_load_bT(lB + j * 16 * 32, 32, 0);
#pragma unroll
        for (int i = 0; i < 4; ++i) {
            v16h a = wmma_load_a(lA + i * 16 * 32, 32, 0);
#pragma unroll
            for (int j = 0; j < 2; ++j)
                acc[i][j] = wmma_f16(a, b[j], acc[i][j]);
        }
    }

    const int lane = threadIdx.x & 31;
#pragma unroll
    for (int i = 0; i < 4; ++i) {
#pragma unroll
        for (int j = 0; j < 2; ++j) {
            const int n     = n0 + j * 16 + (lane & 15);
            const int mbase = m0 + i * 16 + ((lane >> 4) << 3);
#pragma unroll
            for (int r = 0; r < 8; ++r)
                C[(size_t)(mbase + r) * N + n] = acc[i][j][r];
        }
    }
}

// ---------------------------------------------------------------------------
// Kernel 2: per-(b,s) rmsnorm(q,k) + RoPE + concat + per-head cat rmsnorm.
// Emits xq,xk as f16 [b][h][s][128] and xv transposed f16 [b][h][d=128][s].
// ---------------------------------------------------------------------------
__global__ void k_qkv_post(const float* __restrict__ comb_lc,
                           const float* __restrict__ comb_gc,
                           const float* __restrict__ freqs,
                           const float* __restrict__ q_lc_nw,
                           const float* __restrict__ k_lc_nw,
                           const float* __restrict__ q_gc_nw,
                           const float* __restrict__ k_gc_nw,
                           const float* __restrict__ q_cat_nw,
                           const float* __restrict__ k_cat_nw,
                           _Float16* __restrict__ xq,
                           _Float16* __restrict__ xk,
                           _Float16* __restrict__ xvT) {
    const int bs = blockIdx.x;           // b*S + s
    const int b  = bs >> 11;
    const int s  = bs & (SS - 1);
    const float* clc = comb_lc + (size_t)bs * TOTALC;
    const float* cgc = comb_gc + (size_t)bs * TOTALC;
    const int t = threadIdx.x;

    __shared__ float red[256];
    __shared__ float qcat[2048];
    __shared__ float kcat[2048];
    __shared__ float hssq[2][NHEADS];
    __shared__ float rmsv[4];

    // --- 4 sum-of-squares reductions over 1024 elements each ---
    float p0 = 0.f, p1 = 0.f, p2 = 0.f, p3 = 0.f;
    for (int i = t; i < QDIM; i += 256) {
        float a = clc[i];          p0 += a * a;     // q_lc
        float c = clc[QDIM + i];   p1 += c * c;     // k_lc
        float d = cgc[i];          p2 += d * d;     // q_gc
        float e = cgc[QDIM + i];   p3 += e * e;     // k_gc
    }
    float part[4] = {p0, p1, p2, p3};
#pragma unroll
    for (int j = 0; j < 4; ++j) {
        red[t] = part[j]; __syncthreads();
        for (int w = 128; w > 0; w >>= 1) {
            if (t < w) red[t] += red[t + w];
            __syncthreads();
        }
        if (t == 0) rmsv[j] = rsqrtf(red[0] * (1.f / QDIM) + EPSV);
        __syncthreads();
    }
    if (t < NHEADS) { hssq[0][t] = 0.f; hssq[1][t] = 0.f; }
    __syncthreads();

    const float rq_lc = rmsv[0], rk_lc = rmsv[1], rq_gc = rmsv[2], rk_gc = rmsv[3];

    // --- build roped/normalized q_cat, k_cat (2048 elems, 8 per thread) ---
    const int e0   = t * 8;
    const int head = e0 >> 7;              // 128 elems per head; 8-chunks stay in-head
    float accq = 0.f, acck = 0.f;
#pragma unroll
    for (int u = 0; u < 8; ++u) {
        const int e      = e0 + u;
        const int within = e & 127;
        const int half   = within >> 6;    // 0 = lc, 1 = gc
        const int j      = within & 63;
        const int i      = j >> 1;
        const float* src = half ? cgc : clc;
        const float  cz  = cosf(freqs[s * (HALFHD / 2) + i]);
        const float  sz  = sinf(freqs[s * (HALFHD / 2) + i]);
        const int    qi  = head * HALFHD + (i << 1);
        // q path
        {
            const float  rr = half ? rq_gc : rq_lc;
            const float* nw = half ? q_gc_nw : q_lc_nw;
            float xr = src[qi]     * rr * nw[qi];
            float xi = src[qi + 1] * rr * nw[qi + 1];
            float v  = (j & 1) ? (xr * sz + xi * cz) : (xr * cz - xi * sz);
            qcat[e] = v; accq += v * v;
        }
        // k path (offset Q_DIM)
        {
            const float  rr = half ? rk_gc : rk_lc;
            const float* nw = half ? k_gc_nw : k_lc_nw;
            float xr = src[QDIM + qi]     * rr * nw[qi];
            float xi = src[QDIM + qi + 1] * rr * nw[qi + 1];
            float v  = (j & 1) ? (xr * sz + xi * cz) : (xr * cz - xi * sz);
            kcat[e] = v; acck += v * v;
        }
    }
    atomicAdd(&hssq[0][head], accq);
    atomicAdd(&hssq[1][head], acck);
    __syncthreads();

    const float rqh = rsqrtf(hssq[0][head] * (1.f / HEADD) + EPSV);
    const float rkh = rsqrtf(hssq[1][head] * (1.f / HEADD) + EPSV);
    const size_t bh = (size_t)(b * NHEADS + head);
#pragma unroll
    for (int u = 0; u < 8; ++u) {
        const int e = e0 + u;
        const int d = e & 127;
        xq[(bh * SS + s) * HEADD + d] = (_Float16)(qcat[e] * rqh * q_cat_nw[d]);
        xk[(bh * SS + s) * HEADD + d] = (_Float16)(kcat[e] * rkh * k_cat_nw[d]);
        const int within = e & 127, half = within >> 6, j = within & 63;
        const float* src = half ? cgc : clc;
        const float v = src[2 * QDIM + head * HALFHD + j];
        xvT[(bh * HEADD + d) * SS + s] = (_Float16)v;
    }
}

// ---------------------------------------------------------------------------
// Kernel 3: attention. One wave per (b, h, 16-query-row tile).
// Scores for all 2048 keys staged in dynamic LDS as f16 (64 KB),
// fp32 softmax, then O = P @ V with 8 WMMA accumulators (16x128 tile).
// ---------------------------------------------------------------------------
__global__ void __launch_bounds__(32, 1)
k_attention(const _Float16* __restrict__ xq,
            const _Float16* __restrict__ xk,
            const _Float16* __restrict__ xvT,
            _Float16* __restrict__ attn_out) {
    extern __shared__ char dynsmem[];
    _Float16* sc = (_Float16*)dynsmem;        // [16][2048] f16 scores
    __shared__ float rowsum[16];

    const int qt = blockIdx.x & ((SS / 16) - 1);     // 128 q-tiles
    const int bh = blockIdx.x >> 7;
    const int b  = bh >> 4;
    const int h  = bh & (NHEADS - 1);
    const int q0 = qt * 16;
    const int lane = threadIdx.x & 31;

    const _Float16* Q = xq  + (size_t)bh * SS * HEADD + (size_t)q0 * HEADD;
    const _Float16* K = xk  + (size_t)bh * SS * HEADD;
    const _Float16* V = xvT + (size_t)bh * HEADD * SS;

    // Query A-fragments for d = 0,32,64,96
    v16h aq[4];
#pragma unroll
    for (int c = 0; c < 4; ++c) aq[c] = wmma_load_a(Q, HEADD, c * 32);

    const float scale = 0.08838834764831845f;  // 1/sqrt(128)

    // ---- scores: S[16 x 2048] ----
    for (int kt = 0; kt < SS / 16; ++kt) {
        v8f acc = {};
#pragma unroll
        for (int c = 0; c < 4; ++c) {
            v16h bk = wmma_load_bT(K + (size_t)(kt * 16) * HEADD, HEADD, c * 32);
            acc = wmma_f16(aq[c], bk, acc);
        }
        const int n     = kt * 16 + (lane & 15);
        const int mbase = (lane >> 4) << 3;
#pragma unroll
        for (int r = 0; r < 8; ++r)
            sc[(mbase + r) * SS + n] = (_Float16)(acc[r] * scale);
    }
    __syncthreads();

    // ---- softmax over 2048 keys, 2 lanes per row ----
    {
        const int row   = lane >> 1;
        const int halfc = lane & 1;
        float mx = -1e30f;
        for (int c = 0; c < SS / 2; ++c)
            mx = fmaxf(mx, (float)sc[row * SS + halfc * (SS / 2) + c]);
        mx = fmaxf(mx, __shfl_xor(mx, 1));
        float sum = 0.f;
        for (int c = 0; c < SS / 2; ++c) {
            const int idx = row * SS + halfc * (SS / 2) + c;
            float v = __expf((float)sc[idx] - mx);
            sum += v;
            sc[idx] = (_Float16)v;              // unnormalized probs
        }
        sum += __shfl_xor(sum, 1);
        if (halfc == 0) rowsum[row] = sum;
    }
    __syncthreads();

    // ---- O[16 x 128] = P @ V (V stored [d][s] so B-frags are contiguous) ----
    v8f oacc[8];
#pragma unroll
    for (int i = 0; i < 8; ++i) oacc[i] = (v8f){};
    for (int k = 0; k < SS; k += 32) {
        v16h ap = wmma_load_a(sc, SS, k);       // probs from LDS
#pragma unroll
        for (int nc = 0; nc < 8; ++nc) {
            v16h bv = wmma_load_bT(V + (size_t)(nc * 16) * SS, SS, k);
            oacc[nc] = wmma_f16(ap, bv, oacc[nc]);
        }
    }

    // ---- epilogue: fold 1/rowsum, store f16 at [b][s][h*128 + d] ----
    const int mbase = (lane >> 4) << 3;
#pragma unroll
    for (int nc = 0; nc < 8; ++nc) {
        const int d = nc * 16 + (lane & 15);
#pragma unroll
        for (int r = 0; r < 8; ++r) {
            const int qrow = mbase + r;
            const float inv = 1.0f / rowsum[qrow];
            attn_out[((size_t)b * SS + q0 + qrow) * DIMC + h * HEADD + d] =
                (_Float16)(oacc[nc][r] * inv);
        }
    }
}

// ---------------------------------------------------------------------------
// Kernel 4: out = rmsnorm(y + bias) * w   per row of 2048, fp32
// ---------------------------------------------------------------------------
__global__ void k_bias_rms(const float* __restrict__ Y,
                           const float* __restrict__ bias,
                           const float* __restrict__ w,
                           float* __restrict__ out) {
    const int row = blockIdx.x;
    const float* y = Y + (size_t)row * DIMC;
    float* o = out + (size_t)row * DIMC;
    __shared__ float red[256];
    const int t = threadIdx.x;
    float ssq = 0.f;
    for (int i = t; i < DIMC; i += 256) {
        const float v = y[i] + bias[i];
        ssq += v * v;
    }
    red[t] = ssq; __syncthreads();
    for (int s = 128; s > 0; s >>= 1) {
        if (t < s) red[t] += red[t + s];
        __syncthreads();
    }
    const float r = rsqrtf(red[0] * (1.f / DIMC) + EPSV);
    for (int i = t; i < DIMC; i += 256)
        o[i] = (y[i] + bias[i]) * r * w[i];
}

// ---------------------------------------------------------------------------
// Launch
// ---------------------------------------------------------------------------
extern "C" void kernel_launch(void* const* d_in, const int* in_sizes, int n_in,
                              void* d_out, int out_size, void* d_ws, size_t ws_size,
                              hipStream_t stream) {
    const float* local_c  = (const float*)d_in[0];
    const float* global_c = (const float*)d_in[1];
    const float* freqs    = (const float*)d_in[2];
    const float* W_lc     = (const float*)d_in[3];
    const float* W_gc     = (const float*)d_in[4];
    const float* q_lc_nw  = (const float*)d_in[5];
    const float* k_lc_nw  = (const float*)d_in[6];
    const float* q_gc_nw  = (const float*)d_in[7];
    const float* k_gc_nw  = (const float*)d_in[8];
    const float* q_cat_nw = (const float*)d_in[9];
    const float* k_cat_nw = (const float*)d_in[10];
    const float* W_lo     = (const float*)d_in[11];
    const float* b_lo     = (const float*)d_in[12];
    const float* lc_on_w  = (const float*)d_in[13];
    const float* W_go     = (const float*)d_in[14];
    const float* b_go     = (const float*)d_in[15];
    const float* gc_on_w  = (const float*)d_in[16];

    const int MROWS = BB * SS;            // 4096
    const int nX  = MROWS * DIMC;         // 8,388,608
    const int nWq = TOTALC * DIMC;        // 6,291,456
    const int nWo = DIMC * DIMC;          // 4,194,304
    const int nC  = MROWS * TOTALC;       // 12,582,912
    const int nH  = BB * NHEADS * SS * HEADD; // 8,388,608

    char* ws = (char*)d_ws;
    size_t off = 0;
    auto take = [&](size_t bytes) -> void* {
        void* p = ws + off;
        off += (bytes + 255) & ~(size_t)255;
        return p;
    };
    _Float16* Xlc16  = (_Float16*)take((size_t)nX  * 2);
    _Float16* Xgc16  = (_Float16*)take((size_t)nX  * 2);
    _Float16* Wlc16  = (_Float16*)take((size_t)nWq * 2);
    _Float16* Wgc16  = (_Float16*)take((size_t)nWq * 2);
    _Float16* Wlo16  = (_Float16*)take((size_t)nWo * 2);
    _Float16* Wgo16  = (_Float16*)take((size_t)nWo * 2);
    float*    comb_lc = (float*)take((size_t)nC * 4);
    float*    comb_gc = (float*)take((size_t)nC * 4);
    _Float16* xq16   = (_Float16*)take((size_t)nH * 2);
    _Float16* xk16   = (_Float16*)take((size_t)nH * 2);
    _Float16* xvT16  = (_Float16*)take((size_t)nH * 2);
    _Float16* attn16 = (_Float16*)take((size_t)nX * 2);
    float*    y_lc   = (float*)take((size_t)nX * 4);
    float*    y_gc   = (float*)take((size_t)nX * 4);

    // 0) f32 -> f16 conversions
    k_cvt_f16<<<(nX  + 255) / 256, 256, 0, stream>>>(local_c,  Xlc16, nX);
    k_cvt_f16<<<(nX  + 255) / 256, 256, 0, stream>>>(global_c, Xgc16, nX);
    k_cvt_f16<<<(nWq + 255) / 256, 256, 0, stream>>>(W_lc, Wlc16, nWq);
    k_cvt_f16<<<(nWq + 255) / 256, 256, 0, stream>>>(W_gc, Wgc16, nWq);
    k_cvt_f16<<<(nWo + 255) / 256, 256, 0, stream>>>(W_lo, Wlo16, nWo);
    k_cvt_f16<<<(nWo + 255) / 256, 256, 0, stream>>>(W_go, Wgo16, nWo);

    const size_t gemm_lds = 4 * 12288;    // 4 waves x 12 KB double-buffered slice

    // 1) QKV projections: comb = X @ W^T  (M=4096, N=3072, K=2048)
    {
        const int tiles = (MROWS / 64) * (TOTALC / 32);      // 6144
        const int grid  = (tiles + 3) / 4;                   // 4 waves/block
        k_gemm_wmma<<<grid, 128, gemm_lds, stream>>>(Xlc16, Wlc16, comb_lc, MROWS, TOTALC, DIMC);
        k_gemm_wmma<<<grid, 128, gemm_lds, stream>>>(Xgc16, Wgc16, comb_gc, MROWS, TOTALC, DIMC);
    }

    // 2) rmsnorm + rope + concat + per-head norm
    k_qkv_post<<<MROWS, 256, 0, stream>>>(comb_lc, comb_gc, freqs,
                                          q_lc_nw, k_lc_nw, q_gc_nw, k_gc_nw,
                                          q_cat_nw, k_cat_nw,
                                          xq16, xk16, xvT16);

    // 3) attention: one wave per (b,h,16-q-rows); 64KB dynamic LDS for scores
    {
        const int grid = BB * NHEADS * (SS / 16);            // 4096
        const size_t lds = (size_t)16 * SS * sizeof(_Float16); // 65536
        k_attention<<<grid, 32, lds, stream>>>(xq16, xk16, xvT16, attn16);
    }

    // 4) output projections (M=4096, N=2048, K=2048)
    {
        const int tiles = (MROWS / 64) * (DIMC / 32);        // 4096
        const int grid  = (tiles + 3) / 4;
        k_gemm_wmma<<<grid, 128, gemm_lds, stream>>>(attn16, Wlo16, y_lc, MROWS, DIMC, DIMC);
        k_gemm_wmma<<<grid, 128, gemm_lds, stream>>>(attn16, Wgo16, y_gc, MROWS, DIMC, DIMC);
    }

    // 5) bias + rmsnorm epilogues -> d_out (x_lc then x_gc, fp32)
    float* out_lc = (float*)d_out;
    float* out_gc = out_lc + (size_t)nX;
    k_bias_rms<<<MROWS, 256, 0, stream>>>(y_lc, b_lo, lc_on_w, out_lc);
    k_bias_rms<<<MROWS, 256, 0, stream>>>(y_gc, b_go, gc_on_w, out_gc);
}
